// Model3_16484084483095
// MI455X (gfx1250) — matrix operand order repaired and checked by reading, not compiled, not run on hardware
//
#include <hip/hip_runtime.h>

// ---------------------------------------------------------------------------
// EdgeConv GNN for MI455X (gfx1250, wave32, WMMA f32_16x16x32_f16)
// ---------------------------------------------------------------------------
#define N_NODES 50000
#define N_EDGES 800000

typedef __attribute__((ext_vector_type(16))) _Float16 v16h;
typedef __attribute__((ext_vector_type(8)))  float    v8f;

// ---- A fragment from LDS (16-bit A 16x32, contiguous K order) ---------------
// lane<16 : row M=lane,    K = k0..k0+15   (k0 = kt*32)
// lane>=16: row M=lane-16, K = k0+16..k0+31
__device__ __forceinline__ v16h lds_a_frag(const _Float16* row, int kt, int lane) {
  const int k0 = kt * 32 + ((lane >> 4) << 4);
  union { uint4 u[2]; v16h h; } t;
  t.u[0] = *reinterpret_cast<const uint4*>(row + k0);
  t.u[1] = *reinterpret_cast<const uint4*>(row + k0 + 8);
  return t.h;
}

// ---- B fragment from pre-packed global (lane-major, 16 halves/lane) --------
__device__ __forceinline__ v16h ld_b_frag(const _Float16* frag, int lane) {
  union { uint4 u[2]; v16h h; } t;
  const uint4* p = reinterpret_cast<const uint4*>(frag + (size_t)lane * 16);
  t.u[0] = p[0];
  t.u[1] = p[1];
  return t.h;
}

// ---- one 16x16 D tile: act[32][ldw] (f16, LDS) @ packed W ------------------
template <int KTILES, int NTILES>
__device__ __forceinline__ v8f tile_mm(const _Float16* act, int ldw, int mt, int nt,
                                       const _Float16* wp, int lane) {
  v8f acc = {0.f, 0.f, 0.f, 0.f, 0.f, 0.f, 0.f, 0.f};
  const _Float16* arow = act + (size_t)(mt * 16 + (lane & 15)) * ldw;
#pragma unroll
  for (int kt = 0; kt < KTILES; ++kt) {
    v16h a = lds_a_frag(arow, kt, lane);
    v16h b = ld_b_frag(wp + (size_t)(kt * NTILES + nt) * 512, lane);
    acc = __builtin_amdgcn_wmma_f32_16x16x32_f16(
        /*neg_a=*/false, a, /*neg_b=*/false, b,
        /*c_mod=*/(short)0, acc, /*reuse_a=*/false, /*reuse_b=*/false);
  }
  return acc;
}

// ---- write a D tile (bias + relu) back to an LDS activation buffer ---------
__device__ __forceinline__ void store_hidden(_Float16* act, int ldw, int mt, int nt,
                                             int lane, v8f acc, const float* bias) {
  const int col = nt * 16 + (lane & 15);
  const float bv = bias[col];
  const int mb = mt * 16 + ((lane >> 4) << 3);
#pragma unroll
  for (int v = 0; v < 8; ++v) {
    float xv = fmaxf(acc[v] + bv, 0.0f);
    act[(size_t)(mb + v) * ldw + col] = (_Float16)xv;
  }
}

// ---- pack 8 floats -> 8 halves -> one ds_store_b128 ------------------------
__device__ __forceinline__ void store8h(_Float16* dst, const float* v) {
  union { uint4 u; _Float16 h[8]; } t;
#pragma unroll
  for (int i = 0; i < 8; ++i) t.h[i] = (_Float16)v[i];
  *reinterpret_cast<uint4*>(dst) = t.u;
}

// ===========================================================================
// workspace zeroing (cnt + agg1 + ctx + agg2, contiguous N*193 floats)
// ===========================================================================
__global__ void k_zero(float* __restrict__ p, long n) {
  long i = (long)blockIdx.x * blockDim.x + threadIdx.x;
  const long stride = (long)gridDim.x * blockDim.x;
  for (; i < n; i += stride) p[i] = 0.0f;
}

// ===========================================================================
// weight packing: fp32 (K x Ncols, row-major) -> WMMA B fragments (f16)
// frag(kt,nt): lane n<16 holds col nt*16+n, K = kt*32+0..15 ; lane+16: K+16..31
// perm194 remaps the 194-dim layer-2 input to [x_i|x_d|ctx|pdiff|pad224]
// ===========================================================================
__global__ void k_pack(const float* __restrict__ W, int realK, int ncols,
                       int ktiles, int ntiles, _Float16* __restrict__ dst,
                       int perm194) {
  const int t = blockIdx.x * blockDim.x + threadIdx.x;
  const int total = ktiles * ntiles * 32;
  if (t >= total) return;
  const int lane = t & 31;
  const int f = t >> 5;
  const int nt = f % ntiles;
  const int kt = f / ntiles;
  const int n = nt * 16 + (lane & 15);
  const int kb = kt * 32 + ((lane >> 4) * 16);
  _Float16 o[16];
#pragma unroll
  for (int i = 0; i < 16; ++i) {
    int k = kb + i;
    int ok = k;
    if (perm194) {
      if (k < 128)       ok = k;          // x_i, x_j - x_i
      else if (k < 192)  ok = k + 2;      // ctx (orig rows 130..193)
      else if (k == 192) ok = 128;        // pdx
      else if (k == 193) ok = 129;        // pdy
      else               ok = -1;         // pad
    }
    const float v = (ok >= 0 && ok < realK) ? W[(size_t)ok * ncols + n] : 0.0f;
    o[i] = (_Float16)v;
  }
  _Float16* d = dst + (size_t)t * 16;
#pragma unroll
  for (int i = 0; i < 16; ++i) d[i] = o[i];
}

// ===========================================================================
// Layer 1: per-edge msg_net(pos_j - pos_i) = 2 ->128(relu)->64, atomic scatter
// 4 waves/block, 32 edges/wave (800000 = 6250 * 128 exactly)
// ===========================================================================
__global__ void __launch_bounds__(128) k_l1_edge(
    const float* __restrict__ pos, const int* __restrict__ ei,
    const float* __restrict__ w1, const float* __restrict__ b1,
    const _Float16* __restrict__ w2p, const float* __restrict__ b2,
    float* __restrict__ agg1, float* __restrict__ cnt) {
  __shared__ __align__(16) _Float16 hbuf[4][32][128];
  __shared__ int rbuf[4][32];
  const int lane = threadIdx.x & 31;
  const int wv = threadIdx.x >> 5;
  const int e = (blockIdx.x * 4 + wv) * 32 + lane;
  const int r = ei[e];
  const int c = ei[N_EDGES + e];
  rbuf[wv][lane] = r;
  const float pdx = pos[2 * c] - pos[2 * r];
  const float pdy = pos[2 * c + 1] - pos[2 * r + 1];
  atomicAdd(&cnt[r], 1.0f);                       // segment counts (once, reused)
  _Float16* hrow = &hbuf[wv][lane][0];
  for (int j = 0; j < 128; j += 8) {              // hidden: K=2, VALU is right
    float t[8];
#pragma unroll
    for (int i = 0; i < 8; ++i)
      t[i] = fmaxf(fmaf(w1[j + i], pdx, fmaf(w1[128 + j + i], pdy, b1[j + i])), 0.0f);
    store8h(hrow + j, t);
  }
  __syncthreads();
  const _Float16* act = &hbuf[wv][0][0];
  for (int mt = 0; mt < 2; ++mt)
    for (int nt = 0; nt < 4; ++nt) {
      v8f acc = tile_mm<4, 4>(act, 128, mt, nt, w2p, lane);   // 128 -> 64
      const int col = nt * 16 + (lane & 15);
      const float bv = b2[col];
      const int mb = mt * 16 + ((lane >> 4) << 3);
#pragma unroll
      for (int v = 0; v < 8; ++v)
        atomicAdd(&agg1[(size_t)rbuf[wv][mb + v] * 64 + col], acc[v] + bv);
    }
}

// ===========================================================================
// ctx = aggr_net(mean(agg1)) : 64 ->128(relu)->64 ; 2 waves/block, 32 nodes/wave
// ===========================================================================
__global__ void __launch_bounds__(64) k_ctx(
    const float* __restrict__ agg1, const float* __restrict__ cnt,
    const _Float16* __restrict__ w1p, const float* __restrict__ b1,
    const _Float16* __restrict__ w2p, const float* __restrict__ b2,
    float* __restrict__ ctx) {
  __shared__ __align__(16) _Float16 abuf[2][32][64];
  __shared__ __align__(16) _Float16 hbuf[2][32][128];
  const int lane = threadIdx.x & 31;
  const int wv = threadIdx.x >> 5;
  const int nb = (blockIdx.x * 2 + wv) * 32;
  int n = nb + lane;
  if (n >= N_NODES) n = N_NODES - 1;              // clamp; guarded on store
  const float inv = 1.0f / fmaxf(cnt[n], 1.0f);
  _Float16* arow = &abuf[wv][lane][0];
  for (int j = 0; j < 64; j += 8) {
    float t[8];
#pragma unroll
    for (int i = 0; i < 8; ++i) t[i] = agg1[(size_t)n * 64 + j + i] * inv;
    store8h(arow + j, t);
  }
  __syncthreads();
  for (int mt = 0; mt < 2; ++mt)
    for (int nt = 0; nt < 8; ++nt) {
      v8f acc = tile_mm<2, 8>(&abuf[wv][0][0], 64, mt, nt, w1p, lane);
      store_hidden(&hbuf[wv][0][0], 128, mt, nt, lane, acc, b1);
    }
  __syncthreads();
  for (int mt = 0; mt < 2; ++mt)
    for (int nt = 0; nt < 4; ++nt) {
      v8f acc = tile_mm<4, 4>(&hbuf[wv][0][0], 128, mt, nt, w2p, lane);
      const int col = nt * 16 + (lane & 15);
      const float bv = b2[col];
      const int mb = mt * 16 + ((lane >> 4) << 3);
#pragma unroll
      for (int v = 0; v < 8; ++v) {
        const int g = nb + mb + v;
        if (g < N_NODES) ctx[(size_t)g * 64 + col] = acc[v] + bv;
      }
    }
}

// ===========================================================================
// Layer 2: msg_net([x_i | x_j - x_i | ctx_i | pdiff | pad]) 224->128(relu)->64
// 2 waves/block, 32 edges/wave (800000 = 12500 * 64 exactly)
// ===========================================================================
__global__ void __launch_bounds__(64) k_l2_edge(
    const float* __restrict__ x, const float* __restrict__ pos,
    const int* __restrict__ ei, const float* __restrict__ ctx,
    const _Float16* __restrict__ w1p, const float* __restrict__ b1,
    const _Float16* __restrict__ w2p, const float* __restrict__ b2,
    float* __restrict__ agg2) {
  __shared__ __align__(16) _Float16 msg[2][32][224];
  __shared__ __align__(16) _Float16 hbuf[2][32][128];
  __shared__ int rbuf[2][32];
  const int lane = threadIdx.x & 31;
  const int wv = threadIdx.x >> 5;
  const int e = (blockIdx.x * 2 + wv) * 32 + lane;
  const int r = ei[e];
  const int c = ei[N_EDGES + e];
  rbuf[wv][lane] = r;
  const float pdx = pos[2 * c] - pos[2 * r];
  const float pdy = pos[2 * c + 1] - pos[2 * r + 1];
  _Float16* mrow = &msg[wv][lane][0];
  const float* xi = x + (size_t)r * 64;
  const float* xj = x + (size_t)c * 64;
  const float* cr = ctx + (size_t)r * 64;
  for (int j = 0; j < 64; j += 8) {               // the memory-bound gather
    float t0[8], t1[8], t2[8];
#pragma unroll
    for (int i = 0; i < 8; ++i) {
      const float a = xi[j + i];
      const float b = xj[j + i];
      t0[i] = a;
      t1[i] = b - a;
      t2[i] = cr[j + i];
    }
    store8h(mrow + j, t0);
    store8h(mrow + 64 + j, t1);
    store8h(mrow + 128 + j, t2);
  }
  {
    float t[8] = {pdx, pdy, 0.f, 0.f, 0.f, 0.f, 0.f, 0.f};
    store8h(mrow + 192, t);
    float z[8] = {0.f, 0.f, 0.f, 0.f, 0.f, 0.f, 0.f, 0.f};
    store8h(mrow + 200, z);
    store8h(mrow + 208, z);
    store8h(mrow + 216, z);
  }
  __syncthreads();
  for (int mt = 0; mt < 2; ++mt)
    for (int nt = 0; nt < 8; ++nt) {
      v8f acc = tile_mm<7, 8>(&msg[wv][0][0], 224, mt, nt, w1p, lane);  // 224->128
      store_hidden(&hbuf[wv][0][0], 128, mt, nt, lane, acc, b1);
    }
  __syncthreads();
  for (int mt = 0; mt < 2; ++mt)
    for (int nt = 0; nt < 4; ++nt) {
      v8f acc = tile_mm<4, 4>(&hbuf[wv][0][0], 128, mt, nt, w2p, lane); // 128->64
      const int col = nt * 16 + (lane & 15);
      const float bv = b2[col];
      const int mb = mt * 16 + ((lane >> 4) << 3);
#pragma unroll
      for (int v = 0; v < 8; ++v)
        atomicAdd(&agg2[(size_t)rbuf[wv][mb + v] * 64 + col], acc[v] + bv);
    }
}

// ===========================================================================
// out = aggr_net([x | mean(agg2)]) : 128 ->128(relu)->64
// ===========================================================================
__global__ void __launch_bounds__(64) k_out(
    const float* __restrict__ x, const float* __restrict__ agg2,
    const float* __restrict__ cnt,
    const _Float16* __restrict__ w1p, const float* __restrict__ b1,
    const _Float16* __restrict__ w2p, const float* __restrict__ b2,
    float* __restrict__ out) {
  __shared__ __align__(16) _Float16 abuf[2][32][128];
  __shared__ __align__(16) _Float16 hbuf[2][32][128];
  const int lane = threadIdx.x & 31;
  const int wv = threadIdx.x >> 5;
  const int nb = (blockIdx.x * 2 + wv) * 32;
  int n = nb + lane;
  if (n >= N_NODES) n = N_NODES - 1;
  const float inv = 1.0f / fmaxf(cnt[n], 1.0f);
  _Float16* arow = &abuf[wv][lane][0];
  for (int j = 0; j < 64; j += 8) {
    float t0[8], t1[8];
#pragma unroll
    for (int i = 0; i < 8; ++i) {
      t0[i] = x[(size_t)n * 64 + j + i];
      t1[i] = agg2[(size_t)n * 64 + j + i] * inv;
    }
    store8h(arow + j, t0);
    store8h(arow + 64 + j, t1);
  }
  __syncthreads();
  for (int mt = 0; mt < 2; ++mt)
    for (int nt = 0; nt < 8; ++nt) {
      v8f acc = tile_mm<4, 8>(&abuf[wv][0][0], 128, mt, nt, w1p, lane);
      store_hidden(&hbuf[wv][0][0], 128, mt, nt, lane, acc, b1);
    }
  __syncthreads();
  for (int mt = 0; mt < 2; ++mt)
    for (int nt = 0; nt < 4; ++nt) {
      v8f acc = tile_mm<4, 4>(&hbuf[wv][0][0], 128, mt, nt, w2p, lane);
      const int col = nt * 16 + (lane & 15);
      const float bv = b2[col];
      const int mb = mt * 16 + ((lane >> 4) << 3);
#pragma unroll
      for (int v = 0; v < 8; ++v) {
        const int g = nb + mb + v;
        if (g < N_NODES) out[(size_t)g * 64 + col] = acc[v] + bv;
      }
    }
}

// ===========================================================================
extern "C" void kernel_launch(void* const* d_in, const int* in_sizes, int n_in,
                              void* d_out, int out_size, void* d_ws, size_t ws_size,
                              hipStream_t stream) {
  (void)in_sizes; (void)n_in; (void)out_size; (void)ws_size;
  const float* x      = (const float*)d_in[0];
  const float* pos    = (const float*)d_in[1];
  const int*   ei     = (const int*)  d_in[2];
  const float* l1m_w1 = (const float*)d_in[3];
  const float* l1m_b1 = (const float*)d_in[4];
  const float* l1m_w2 = (const float*)d_in[5];
  const float* l1m_b2 = (const float*)d_in[6];
  const float* l1a_w1 = (const float*)d_in[7];
  const float* l1a_b1 = (const float*)d_in[8];
  const float* l1a_w2 = (const float*)d_in[9];
  const float* l1a_b2 = (const float*)d_in[10];
  const float* l2m_w1 = (const float*)d_in[11];
  const float* l2m_b1 = (const float*)d_in[12];
  const float* l2m_w2 = (const float*)d_in[13];
  const float* l2m_b2 = (const float*)d_in[14];
  const float* l2a_w1 = (const float*)d_in[15];
  const float* l2a_b1 = (const float*)d_in[16];
  const float* l2a_w2 = (const float*)d_in[17];
  const float* l2a_b2 = (const float*)d_in[18];

  // workspace layout (floats, then 16B-aligned f16 packed weights)
  float* ws   = (float*)d_ws;
  float* cnt  = ws;                                 // [N]
  float* agg1 = cnt + N_NODES;                      // [N,64]
  float* ctx  = agg1 + (size_t)N_NODES * 64;        // [N,64]
  float* agg2 = ctx  + (size_t)N_NODES * 64;        // [N,64]
  _Float16* wp = (_Float16*)(agg2 + (size_t)N_NODES * 64);
  _Float16* P_l1m_w2 = wp;                          // kt4  nt4  ->  8192
  _Float16* P_l1a_w1 = P_l1m_w2 + 8192;             // kt2  nt8  ->  8192
  _Float16* P_l1a_w2 = P_l1a_w1 + 8192;             // kt4  nt4  ->  8192
  _Float16* P_l2m_w1 = P_l1a_w2 + 8192;             // kt7  nt8  -> 28672
  _Float16* P_l2m_w2 = P_l2m_w1 + 28672;            // kt4  nt4  ->  8192
  _Float16* P_l2a_w1 = P_l2m_w2 + 8192;             // kt4  nt8  -> 16384
  _Float16* P_l2a_w2 = P_l2a_w1 + 16384;            // kt4  nt4  ->  8192

  // 1) zero accumulators (deterministic per call)
  k_zero<<<4096, 256, 0, stream>>>(ws, (long)N_NODES * 193);

  // 2) pack weights into WMMA B fragments
  auto packGrid = [](int kt, int nt) { return (kt * nt * 32 + 255) / 256; };
  k_pack<<<packGrid(4, 4), 256, 0, stream>>>(l1m_w2, 128, 64,  4, 4, P_l1m_w2, 0);
  k_pack<<<packGrid(2, 8), 256, 0, stream>>>(l1a_w1, 64, 128,  2, 8, P_l1a_w1, 0);
  k_pack<<<packGrid(4, 4), 256, 0, stream>>>(l1a_w2, 128, 64,  4, 4, P_l1a_w2, 0);
  k_pack<<<packGrid(7, 8), 256, 0, stream>>>(l2m_w1, 194, 128, 7, 8, P_l2m_w1, 1);
  k_pack<<<packGrid(4, 4), 256, 0, stream>>>(l2m_w2, 128, 64,  4, 4, P_l2m_w2, 0);
  k_pack<<<packGrid(4, 8), 256, 0, stream>>>(l2a_w1, 128, 128, 4, 8, P_l2a_w1, 0);
  k_pack<<<packGrid(4, 4), 256, 0, stream>>>(l2a_w2, 128, 64,  4, 4, P_l2a_w2, 0);

  // 3) layer-1 edge messages + counts + scatter      (800000 = 6250*128)
  k_l1_edge<<<6250, 128, 0, stream>>>(pos, ei, l1m_w1, l1m_b1, P_l1m_w2, l1m_b2,
                                      agg1, cnt);
  // 4) node context MLP
  k_ctx<<<(N_NODES + 63) / 64, 64, 0, stream>>>(agg1, cnt, P_l1a_w1, l1a_b1,
                                                P_l1a_w2, l1a_b2, ctx);
  // 5) layer-2 edge messages + scatter               (800000 = 12500*64)
  k_l2_edge<<<12500, 64, 0, stream>>>(x, pos, ei, ctx, P_l2m_w1, l2m_b1,
                                      P_l2m_w2, l2m_b2, agg2);
  // 6) output node MLP
  k_out<<<(N_NODES + 63) / 64, 64, 0, stream>>>(x, agg2, cnt, P_l2a_w1, l2a_b1,
                                                P_l2a_w2, l2a_b2, (float*)d_out);
}
